// PtrNet_74182675137017
// MI455X (gfx1250) — matrix operand order repaired and checked by reading, not compiled, not run on hardware
//
#include <hip/hip_runtime.h>

// ---------------------------------------------------------------------------
// Pointer-network (encoder/decoder LSTM + attention) for MI455X (gfx1250).
// All GEMMs via V_WMMA_F32_16X16X32_BF16 (bf16 inputs, f32 accumulate).
// Dims: S=50, B=128, I=128, H=256, 4H=1024.
// ---------------------------------------------------------------------------

typedef __attribute__((ext_vector_type(16))) __bf16 v16bf;
typedef __attribute__((ext_vector_type(8)))  __bf16 v8bf;
typedef __attribute__((ext_vector_type(8)))  float  v8f;

#define SEQ  50
#define BSZ  128
#define IIN  128
#define HH   256
#define G4H  1024

__device__ __forceinline__ float sigmoidf_(float x) {
    return 1.0f / (1.0f + __expf(-x));
}

__device__ __forceinline__ v8f wmma_bf16(v16bf a, v16bf b, v8f c) {
    return __builtin_amdgcn_wmma_f32_16x16x32_bf16(
        /*neg_a=*/false, a, /*neg_b=*/false, b,
        /*c_mod=*/(short)0, c, /*reuse_a=*/false, /*reuse_b=*/false);
}

// A fragment (16x32, 16-bit layout): lane<16 -> row m, K {kb..kb+7, kb+16..kb+23}
//                                    lane>=16 -> same row space, K offset +8.
// Source is row-major f32 [*, lda]; convert to bf16 on load.
__device__ __forceinline__ v16bf a_frag_from_f32(const float* __restrict__ A,
                                                 int lda, int mbase, int kbase,
                                                 int lane) {
    int m = mbase + (lane & 15);
    int k = kbase + ((lane >> 4) << 3);
    const float* p = A + (size_t)m * lda + k;
    v16bf r;
#pragma unroll
    for (int i = 0; i < 8; ++i) r[i]     = (__bf16)p[i];
#pragma unroll
    for (int i = 0; i < 8; ++i) r[8 + i] = (__bf16)p[16 + i];
    return r;
}

// B fragment (32x16, 16-bit layout): lane&15 = N column, K {kb..kb+7, kb+16..kb+23}, lanes>=16 K offset +8.
// Weight stored row-major [N, K] in bf16 -> two 16B loads per lane.
__device__ __forceinline__ v16bf b_frag_from_bf16(const __bf16* __restrict__ W,
                                                  int ldw, int nbase, int kbase,
                                                  int lane) {
    int n = nbase + (lane & 15);
    int k = kbase + ((lane >> 4) << 3);
    const __bf16* p = W + (size_t)n * ldw + k;
    v8bf lo = *(const v8bf*)(p);
    v8bf hi = *(const v8bf*)(p + 16);
    return __builtin_shufflevector(lo, hi, 0, 1, 2, 3, 4, 5, 6, 7,
                                   8, 9, 10, 11, 12, 13, 14, 15);
}

// ---------------------------------------------------------------------------
// Prep kernels
// ---------------------------------------------------------------------------
__global__ void cvt_bf16_kernel(const float* __restrict__ s,
                                __bf16* __restrict__ d, long n) {
    long i  = (long)blockIdx.x * blockDim.x + threadIdx.x;
    long st = (long)gridDim.x * blockDim.x;
    for (; i < n; i += st) d[i] = (__bf16)s[i];
}

__global__ void add2_kernel(const float* __restrict__ a,
                            const float* __restrict__ b,
                            float* __restrict__ o, int n) {
    int i = blockIdx.x * blockDim.x + threadIdx.x;
    if (i < n) o[i] = a[i] + b[i];
}

// decoder input is constant -1 => x@Wih^T row j = -sum_i Wih[j][i]
__global__ void dec_g0_kernel(const float* __restrict__ Wih,
                              const float* __restrict__ bih,
                              const float* __restrict__ bhh,
                              float* __restrict__ g0, int total, int I) {
    int j = blockIdx.x * blockDim.x + threadIdx.x;
    if (j >= total) return;
    const float* row = Wih + (size_t)j * I;
    float s = 0.f;
    for (int i = 0; i < I; ++i) s += row[i];
    g0[j] = bih[j] + bhh[j] - s;
}

// ---------------------------------------------------------------------------
// Fused LSTM step:  G = x@Wih^T + h@Whh^T + gbias ; cell update ; h,c out.
// Wave computes one 16(M)x16(N) tile for ALL FOUR gates (A-frag reused 4x).
// Grid: 16 blocks x 256 thr (8 waves) = 128 waves = 8 Mtiles x 16 Ntiles.
// x/Wih may be null (decoder: I==0, x-term folded into gbias).
// ---------------------------------------------------------------------------
__global__ __launch_bounds__(256) void lstm_step_kernel(
    const float* __restrict__ x, const __bf16* __restrict__ Wih,
    const __bf16* __restrict__ Whh, const float* __restrict__ gbias,
    const float* __restrict__ h_in, const float* __restrict__ c_in,
    float* __restrict__ h_out, float* __restrict__ c_out,
    float* __restrict__ c_seq, int I) {
    int lane = threadIdx.x & 31;
    int wave = blockIdx.x * 8 + (threadIdx.x >> 5);
    int mt = (wave & 7) << 4;
    int nt = (wave >> 3) << 4;

    v8f z = {};
    v8f acc[4] = {z, z, z, z};

    for (int k = 0; k < I; k += 32) {              // input GEMM (encoder only)
        v16bf a = a_frag_from_f32(x, I, mt, k, lane);
#pragma unroll
        for (int g = 0; g < 4; ++g) {
            v16bf b = b_frag_from_bf16(Wih, I, g * HH + nt, k, lane);
            acc[g] = wmma_bf16(a, b, acc[g]);
        }
    }
    for (int k = 0; k < HH; k += 32) {             // recurrent GEMM
        v16bf a = a_frag_from_f32(h_in, HH, mt, k, lane);
#pragma unroll
        for (int g = 0; g < 4; ++g) {
            v16bf b = b_frag_from_bf16(Whh, HH, g * HH + nt, k, lane);
            acc[g] = wmma_bf16(a, b, acc[g]);
        }
    }

    // epilogue: C/D layout -> M = e + 8*(lane>=16), N = lane&15
    int n    = nt + (lane & 15);
    int mrow = mt + ((lane >> 4) << 3);
    float bi = gbias[n];
    float bf = gbias[HH + n];
    float bg = gbias[2 * HH + n];
    float bo = gbias[3 * HH + n];
#pragma unroll
    for (int e = 0; e < 8; ++e) {
        size_t idx = (size_t)(mrow + e) * HH + n;
        float gi = sigmoidf_(acc[0][e] + bi);
        float gf = sigmoidf_(acc[1][e] + bf);
        float gg = tanhf(acc[2][e] + bg);
        float go = sigmoidf_(acc[3][e] + bo);
        float cn = gf * c_in[idx] + gi * gg;
        float hn = go * tanhf(cn);
        c_out[idx] = cn;
        h_out[idx] = hn;
        if (c_seq) c_seq[idx] = cn;
    }
}

// ---------------------------------------------------------------------------
// Generic  out[M,N] = A[M,K](f32) @ W[N,K](bf16)^T + bias   (E1 and c1)
// ---------------------------------------------------------------------------
__global__ __launch_bounds__(256) void gemm_bias_kernel(
    const float* __restrict__ A, const __bf16* __restrict__ W,
    const float* __restrict__ bias, float* __restrict__ out,
    int Mtiles, int K, int N) {
    int lane = threadIdx.x & 31;
    int wave = blockIdx.x * 8 + (threadIdx.x >> 5);
    int mt = (wave % Mtiles) << 4;
    int nt = (wave / Mtiles) << 4;

    v8f acc = {};
    for (int k = 0; k < K; k += 32) {
        v16bf a = a_frag_from_f32(A, K, mt, k, lane);
        v16bf b = b_frag_from_bf16(W, K, nt, k, lane);
        acc = wmma_bf16(a, b, acc);
    }
    int n    = nt + (lane & 15);
    int mrow = mt + ((lane >> 4) << 3);
    float bv = bias[n];
#pragma unroll
    for (int e = 0; e < 8; ++e)
        out[(size_t)(mrow + e) * N + n] = acc[e] + bv;
}

// ---------------------------------------------------------------------------
// Attention:  u[sd,se,b,n] = tanh(c1[sd]+E1[se])[b,:] @ V^T + Vb
// One block per (sd, se, 16-row M tile). The tanh A-tile (16x256) is built
// once in LDS (bf16, padded stride 264 = bank-conflict free), then 8 waves
// each produce two 16x16 N tiles (K = 256 -> 8 WMMA chunks per tile).
// ---------------------------------------------------------------------------
__global__ __launch_bounds__(256) void attn_kernel(
    const float* __restrict__ c1, const float* __restrict__ E1,
    const __bf16* __restrict__ V, const float* __restrict__ Vb,
    float* __restrict__ u) {
    constexpr int LDA = 264;  // 256 + 8 pad (bf16 elems)
    __shared__ __attribute__((aligned(16))) __bf16 As[16 * LDA];

    int bx    = blockIdx.x;           // = ((sd*50)+se)*8 + mtIdx
    int mtIdx = bx & 7;
    int pair  = bx >> 3;
    int se    = pair % SEQ;
    int sd    = pair / SEQ;
    int tid   = threadIdx.x;

    const float* c1p = c1 + ((size_t)sd * BSZ + mtIdx * 16) * HH;
    const float* E1p = E1 + ((size_t)se * BSZ + mtIdx * 16) * HH;
#pragma unroll 4
    for (int idx = tid; idx < 16 * HH; idx += 256) {
        int r = idx >> 8;          // HH == 256
        int k = idx & (HH - 1);
        As[r * LDA + k] = (__bf16)tanhf(c1p[(size_t)r * HH + k] +
                                        E1p[(size_t)r * HH + k]);
    }
    __syncthreads();

    int lane = tid & 31;
    int wave = tid >> 5;
    int nt0  = wave << 5;          // this wave: N tiles nt0 and nt0+16

    v8f acc0 = {}, acc1 = {};
    const __bf16* arow = As + (lane & 15) * LDA + ((lane >> 4) << 3);
    for (int k = 0; k < HH; k += 32) {
        v8bf lo = *(const v8bf*)(arow + k);
        v8bf hi = *(const v8bf*)(arow + k + 16);
        v16bf a = __builtin_shufflevector(lo, hi, 0, 1, 2, 3, 4, 5, 6, 7,
                                          8, 9, 10, 11, 12, 13, 14, 15);
        v16bf b0 = b_frag_from_bf16(V, HH, nt0, k, lane);
        v16bf b1 = b_frag_from_bf16(V, HH, nt0 + 16, k, lane);
        acc0 = wmma_bf16(a, b0, acc0);
        acc1 = wmma_bf16(a, b1, acc1);
    }

    size_t base = (((size_t)sd * SEQ + se) * BSZ + mtIdx * 16) * HH;
    int mrow = (lane >> 4) << 3;
    int nc   = lane & 15;
    float bv0 = Vb[nt0 + nc];
    float bv1 = Vb[nt0 + 16 + nc];
#pragma unroll
    for (int e = 0; e < 8; ++e) {
        u[base + (size_t)(mrow + e) * HH + nt0 + nc]      = acc0[e] + bv0;
        u[base + (size_t)(mrow + e) * HH + nt0 + 16 + nc] = acc1[e] + bv1;
    }
}

// ---------------------------------------------------------------------------
// Host orchestration
// ---------------------------------------------------------------------------
extern "C" void kernel_launch(void* const* d_in, const int* in_sizes, int n_in,
                              void* d_out, int out_size, void* d_ws,
                              size_t ws_size, hipStream_t stream) {
    (void)in_sizes; (void)n_in; (void)out_size; (void)ws_size;

    const float* x        = (const float*)d_in[0];   // [50,128,128]
    const float* h0       = (const float*)d_in[1];   // [128,256]
    const float* c0       = (const float*)d_in[2];
    const float* enc_Wih  = (const float*)d_in[3];   // [50,1024,128]
    const float* enc_Whh  = (const float*)d_in[4];   // [50,1024,256]
    const float* enc_bih  = (const float*)d_in[5];   // [50,1024]
    const float* enc_bhh  = (const float*)d_in[6];
    const float* dec_Wih  = (const float*)d_in[7];
    const float* dec_Whh  = (const float*)d_in[8];
    const float* dec_bih  = (const float*)d_in[9];
    const float* dec_bhh  = (const float*)d_in[10];
    const float* We_w     = (const float*)d_in[11];  // [256,256]
    const float* We_b     = (const float*)d_in[12];
    const float* Wd_w     = (const float*)d_in[13];
    const float* Wd_b     = (const float*)d_in[14];
    const float* V_w      = (const float*)d_in[15];
    const float* V_b      = (const float*)d_in[16];
    float* u_out = (float*)d_out;                    // [50,50,128,256] f32

    // ---- workspace carve-up (256B aligned) ----
    char* ws = (char*)d_ws;
    size_t off = 0;
    auto take = [&](size_t bytes) -> char* {
        char* p = ws + off;
        off += (bytes + 255) & ~(size_t)255;
        return p;
    };
    const size_t nWih = (size_t)SEQ * G4H * IIN;   // 6,553,600
    const size_t nWhh = (size_t)SEQ * G4H * HH;    // 13,107,200
    const size_t nSBH = (size_t)SEQ * BSZ * HH;    // 1,638,400

    __bf16* encWih_bf = (__bf16*)take(nWih * 2);
    __bf16* encWhh_bf = (__bf16*)take(nWhh * 2);
    __bf16* decWhh_bf = (__bf16*)take(nWhh * 2);
    __bf16* We_bf     = (__bf16*)take((size_t)HH * HH * 2);
    __bf16* Wd_bf     = (__bf16*)take((size_t)HH * HH * 2);
    __bf16* V_bf      = (__bf16*)take((size_t)HH * HH * 2);
    float* encb   = (float*)take((size_t)SEQ * G4H * 4);
    float* decg0  = (float*)take((size_t)SEQ * G4H * 4);
    float* hbuf   = (float*)take(2 * (size_t)BSZ * HH * 4);  // ping-pong
    float* cbuf   = (float*)take(2 * (size_t)BSZ * HH * 4);
    float* enc_c  = (float*)take(nSBH * 4);
    float* dec_c  = (float*)take(nSBH * 4);
    float* E1     = (float*)take(nSBH * 4);
    float* c1     = (float*)take(nSBH * 4);

    // ---- prep: bf16 weight conversion + fused biases ----
    cvt_bf16_kernel<<<2048, 256, 0, stream>>>(enc_Wih, encWih_bf, (long)nWih);
    cvt_bf16_kernel<<<2048, 256, 0, stream>>>(enc_Whh, encWhh_bf, (long)nWhh);
    cvt_bf16_kernel<<<2048, 256, 0, stream>>>(dec_Whh, decWhh_bf, (long)nWhh);
    cvt_bf16_kernel<<<256, 256, 0, stream>>>(We_w, We_bf, (long)HH * HH);
    cvt_bf16_kernel<<<256, 256, 0, stream>>>(Wd_w, Wd_bf, (long)HH * HH);
    cvt_bf16_kernel<<<256, 256, 0, stream>>>(V_w, V_bf, (long)HH * HH);
    add2_kernel<<<200, 256, 0, stream>>>(enc_bih, enc_bhh, encb, SEQ * G4H);
    dec_g0_kernel<<<200, 256, 0, stream>>>(dec_Wih, dec_bih, dec_bhh, decg0,
                                           SEQ * G4H, IIN);

    // ---- seed recurrent state ----
    hipMemcpyAsync(hbuf, h0, (size_t)BSZ * HH * 4, hipMemcpyDeviceToDevice, stream);
    hipMemcpyAsync(cbuf, c0, (size_t)BSZ * HH * 4, hipMemcpyDeviceToDevice, stream);

    const size_t st = (size_t)BSZ * HH;  // 32768
    int cur = 0;

    // ---- encoder: 50 sequential fused LSTM steps ----
    for (int s = 0; s < SEQ; ++s) {
        int nxt = cur ^ 1;
        lstm_step_kernel<<<16, 256, 0, stream>>>(
            x + (size_t)s * BSZ * IIN,
            encWih_bf + (size_t)s * G4H * IIN,
            encWhh_bf + (size_t)s * G4H * HH,
            encb + (size_t)s * G4H,
            hbuf + cur * st, cbuf + cur * st,
            hbuf + nxt * st, cbuf + nxt * st,
            enc_c + (size_t)s * st, IIN);
        cur = nxt;
    }

    // ---- E1 = enc_c @ We^T + We_b   ([6400,256] x [256,256]) ----
    gemm_bias_kernel<<<800, 256, 0, stream>>>(enc_c, We_bf, We_b, E1,
                                              /*Mtiles=*/400, HH, HH);

    // ---- decoder recurrence (x term folded into decg0) ----
    for (int s = 0; s < SEQ; ++s) {
        int nxt = cur ^ 1;
        lstm_step_kernel<<<16, 256, 0, stream>>>(
            nullptr, nullptr,
            decWhh_bf + (size_t)s * G4H * HH,
            decg0 + (size_t)s * G4H,
            hbuf + cur * st, cbuf + cur * st,
            hbuf + nxt * st, cbuf + nxt * st,
            dec_c + (size_t)s * st, /*I=*/0);
        cur = nxt;
    }

    // ---- c1 = dec_c @ Wd^T + Wd_b ----
    gemm_bias_kernel<<<800, 256, 0, stream>>>(dec_c, Wd_bf, Wd_b, c1,
                                              /*Mtiles=*/400, HH, HH);

    // ---- attention: 50*50*8 = 20000 blocks, dominant 42 GFLOP ----
    attn_kernel<<<SEQ * SEQ * 8, 256, 0, stream>>>(c1, E1, V_bf, V_b, u_out);
}
